// DenseCLHead_37263136260582
// MI455X (gfx1250) — matrix operand order repaired and checked by reading, not compiled, not run on hardware
//
#include <hip/hip_runtime.h>
#include <hip/hip_bf16.h>

typedef __attribute__((ext_vector_type(16))) __bf16 v16bf;
typedef __attribute__((ext_vector_type(8)))  __bf16 v8bf;
typedef __attribute__((ext_vector_type(8)))  float  v8f;

#define BB   64
#define CF   2048
#define HW   196
#define HWP  224      // 7 super-tiles of 32 (14 tiles of 16)
#define CD   256
#define HID  2048
#define NTI  14       // 16-wide position tiles
#define NST  7        // 32-wide position super-tiles

// ---------------------------------------------------------------------------
// Fragment loaders for v_wmma_f32_16x16x32_bf16 (wave32).
// A (16x32, MxK): lane m / m+16 holds row m; K chunks [8h..8h+7], [16+8h..+7].
// B (32x16, KxN): lane n / n+16 holds column n; K chunks [8h..8h+7], [16+8h..+7].
// Both loaders expect row-major / K-contiguous storage.
// ---------------------------------------------------------------------------
__device__ __forceinline__ v16bf load_fragA(const __bf16* rowptr, int k) {
    v8bf lo = *(const v8bf*)(rowptr + k);
    v8bf hi = *(const v8bf*)(rowptr + k + 16);
    v16bf r;
#pragma unroll
    for (int i = 0; i < 8; ++i) { r[i] = lo[i]; r[i + 8] = hi[i]; }
    return r;
}

__device__ __forceinline__ v16bf load_fragB(const __bf16* colptr, int k) {
    v8bf lo = *(const v8bf*)(colptr + k);
    v8bf hi = *(const v8bf*)(colptr + k + 16);
    v16bf r;
#pragma unroll
    for (int i = 0; i < 8; ++i) { r[i] = lo[i]; r[i + 8] = hi[i]; }
    return r;
}

__device__ __forceinline__ v8f wmma_bf16(v16bf a, v16bf b, v8f c) {
    return __builtin_amdgcn_wmma_f32_16x16x32_bf16(false, a, false, b,
                                                   (short)0, c, false, false);
}

// ---------------------------------------------------------------------------
// 0) Weights -> bf16 (row-major, K contiguous: already WMMA A layouts).
// ---------------------------------------------------------------------------
__global__ void k_cvt_w(const float* __restrict__ W1, const float* __restrict__ W2,
                        __bf16* __restrict__ W1b, __bf16* __restrict__ W2b) {
    int i = blockIdx.x * blockDim.x + threadIdx.x;
    if (i < HID * CD) {
        W1b[i] = (__bf16)W1[i];
        W2b[i] = (__bf16)W2[i];
    }
}

// ---------------------------------------------------------------------------
// 1) Per-position inverse L2 norms over CF channels (coalesced along hw).
// ---------------------------------------------------------------------------
__global__ void k_colnorm(const float* __restrict__ f1, const float* __restrict__ f2,
                          float* __restrict__ inv1, float* __restrict__ inv2) {
    int b = blockIdx.x;
    int t = threadIdx.x;
    if (t >= HW) return;
    const float* p1 = f1 + (size_t)b * CF * HW + t;
    const float* p2 = f2 + (size_t)b * CF * HW + t;
    float s1 = 0.f, s2 = 0.f;
    for (int c = 0; c < CF; ++c) {
        float a = p1[(size_t)c * HW]; s1 += a * a;
        float d = p2[(size_t)c * HW]; s2 += d * d;
    }
    inv1[b * HW + t] = 1.f / fmaxf(sqrtf(s1), 1e-8f);
    inv2[b * HW + t] = 1.f / fmaxf(sqrtf(s2), 1e-8f);
}

// ---------------------------------------------------------------------------
// 2) Transpose+normalize feats: (b, CF, HW) f32 -> (b, HWP, CF) bf16,
//    padded rows zero.  16x16 LDS tile transpose.
// ---------------------------------------------------------------------------
__global__ void k_transpose_feat(const float* __restrict__ f1, const float* __restrict__ f2,
                                 const float* __restrict__ inv1, const float* __restrict__ inv2,
                                 __bf16* __restrict__ A1, __bf16* __restrict__ A2) {
    __shared__ float tile[16][17];
    int bb = blockIdx.y, pt = blockIdx.x;
    int x = threadIdx.x, y = threadIdx.y;
    int p0 = pt * 16;
    for (int src = 0; src < 2; ++src) {
        const float* f   = src ? f2   : f1;
        const float* inv = src ? inv2 : inv1;
        __bf16*      A   = src ? A2   : A1;
        for (int ct = 0; ct < CF / 16; ++ct) {
            int c = ct * 16 + y, p = p0 + x;
            tile[y][x] = (p < HW) ? f[((size_t)bb * CF + c) * HW + p] : 0.f;
            __syncthreads();
            int pw = p0 + y, cw = ct * 16 + x;
            float scale = (pw < HW) ? inv[bb * HW + pw] : 0.f;
            A[((size_t)bb * HWP + pw) * CF + cw] = (__bf16)(tile[x][y] * scale);
            __syncthreads();
        }
    }
}

// ---------------------------------------------------------------------------
// 3) dense_on -> xT bf16 (b, HWP, CD); dense_targ -> d2T f32 (b, HW, CD).
// ---------------------------------------------------------------------------
__global__ void k_prep_dense(const float* __restrict__ don, const float* __restrict__ dtg,
                             __bf16* __restrict__ xT, float* __restrict__ d2T) {
    __shared__ float tile[16][17];
    int bb = blockIdx.y, pt = blockIdx.x;
    int x = threadIdx.x, y = threadIdx.y;
    int p0 = pt * 16;
    for (int ct = 0; ct < CD / 16; ++ct) {
        int c = ct * 16 + y, p = p0 + x;
        int pw = p0 + y, cw = ct * 16 + x;
        tile[y][x] = (p < HW) ? don[((size_t)bb * CD + c) * HW + p] : 0.f;
        __syncthreads();
        xT[((size_t)bb * HWP + pw) * CD + cw] = (__bf16)tile[x][y];
        __syncthreads();
        tile[y][x] = (p < HW) ? dtg[((size_t)bb * CD + c) * HW + p] : 0.f;
        __syncthreads();
        if (pw < HW) d2T[((size_t)bb * HW + pw) * CD + cw] = tile[x][y];
        __syncthreads();
    }
}

// ---------------------------------------------------------------------------
// 4) cos_sim_feat = f1n^T * f2n.  One wave per 32x32 super-tile:
//    4 accumulators, 2 A-frags + 2 B-frags per k-step -> 1 KB/WMMA of traffic.
// ---------------------------------------------------------------------------
__global__ __launch_bounds__(256) void k_gemm_cos(const __bf16* __restrict__ A1,
                                                  const __bf16* __restrict__ A2,
                                                  float* __restrict__ cosm) {
    int lane = threadIdx.x & 31;
    int gw   = blockIdx.x * 8 + (threadIdx.x >> 5);   // 64*7*7 = 3136 super-tiles
    int bb = gw / (NST * NST);
    int rem = gw % (NST * NST);
    int it = rem / NST, jt = rem % NST;
    int half = lane >> 4, m = lane & 15;

    const __bf16* ap0 = A1 + ((size_t)bb * HWP + it * 32 + m) * CF + half * 8;
    const __bf16* ap1 = ap0 + (size_t)16 * CF;
    const __bf16* bp0 = A2 + ((size_t)bb * HWP + jt * 32 + m) * CF + half * 8;
    const __bf16* bp1 = bp0 + (size_t)16 * CF;

    v8f c00 = {}, c01 = {}, c10 = {}, c11 = {};
    for (int k = 0; k < CF; k += 32) {
        __builtin_prefetch(ap0 + k + 512, 0, 0);
        v16bf a0 = load_fragA(ap0, k);
        v16bf a1 = load_fragA(ap1, k);
        v16bf b0 = load_fragB(bp0, k);
        v16bf b1 = load_fragB(bp1, k);
        c00 = wmma_bf16(a0, b0, c00);
        c01 = wmma_bf16(a0, b1, c01);
        c10 = wmma_bf16(a1, b0, c10);
        c11 = wmma_bf16(a1, b1, c11);
    }
    // D layout: VGPR r -> row base + r + 8*half, col base + m
    float* o00 = cosm + ((size_t)bb * HWP + it * 32 + half * 8) * HWP + jt * 32 + m;
#pragma unroll
    for (int r = 0; r < 8; ++r) {
        o00[(size_t)r * HWP]             = c00[r];
        o00[(size_t)r * HWP + 16]        = c01[r];
        o00[(size_t)(r + 16) * HWP]      = c10[r];
        o00[(size_t)(r + 16) * HWP + 16] = c11[r];
    }
}

// ---------------------------------------------------------------------------
// 5) argmax over target positions (j < HW) + gather matched dense_targ row.
// ---------------------------------------------------------------------------
__global__ void k_argmax_gather(const float* __restrict__ cosm, const float* __restrict__ d2T,
                                float* __restrict__ matched) {
    int lane = threadIdx.x & 31;
    int row  = blockIdx.x * 8 + (threadIdx.x >> 5);   // 64*196 rows
    int bb = row / HW, i = row % HW;
    const float* crow = cosm + ((size_t)bb * HWP + i) * HWP;
    float best = -3.0e38f; int bidx = 0;
    for (int j = lane; j < HW; j += 32) {
        float v = crow[j];
        if (v > best) { best = v; bidx = j; }
    }
#pragma unroll
    for (int off = 16; off > 0; off >>= 1) {
        float ov = __shfl_xor(best, off, 32);
        int   oi = __shfl_xor(bidx, off, 32);
        if (ov > best || (ov == best && oi < bidx)) { best = ov; bidx = oi; }
    }
    const float* src = d2T     + ((size_t)bb * HW + bidx) * CD;
    float*       dst = matched + ((size_t)bb * HW + i)    * CD;
    for (int c = lane; c < CD; c += 32) dst[c] = src[c];
}

// ---------------------------------------------------------------------------
// 6) Fused 2-layer MLP, one (batch, 16-position) tile per block.
//    Stage 1: H = relu(W1*x + b1) -> 16x2048 bf16 in LDS (64 KB of 320 KB).
//    Stage 2: pred = W2*H + b2.  Both stages pair two M-tiles per shared
//    B-fragment (halves global/LDS B traffic).
// ---------------------------------------------------------------------------
__global__ __launch_bounds__(256) void k_mlp(const __bf16* __restrict__ W1b,
                                             const float* __restrict__ b1,
                                             const __bf16* __restrict__ W2b,
                                             const float* __restrict__ b2,
                                             const __bf16* __restrict__ xT,
                                             float* __restrict__ pred) {
    __shared__ __bf16 Hlds[16 * HID];
    int bb = blockIdx.y, pt = blockIdx.x;
    int lane = threadIdx.x & 31, widx = threadIdx.x >> 5;
    int half = lane >> 4, m = lane & 15;
    int p0 = pt * 16;

    // Stage 1: 128 M-tiles of 16 hid rows, pairs of tiles share the B frag.
    const __bf16* bcol = xT + ((size_t)bb * HWP + p0 + m) * CD + half * 8;
    for (int mt = 2 * widx; mt < HID / 16; mt += 16) {
        const __bf16* arow0 = W1b + (size_t)(mt * 16 + m) * CD + half * 8;
        const __bf16* arow1 = arow0 + (size_t)16 * CD;
        v8f acc0 = {}, acc1 = {};
        for (int k = 0; k < CD; k += 32) {
            v16bf b  = load_fragB(bcol, k);
            v16bf a0 = load_fragA(arow0, k);
            v16bf a1 = load_fragA(arow1, k);
            acc0 = wmma_bf16(a0, b, acc0);
            acc1 = wmma_bf16(a1, b, acc1);
        }
#pragma unroll
        for (int t = 0; t < 2; ++t) {
            v8f acc = t ? acc1 : acc0;
            const float* b1p = b1 + (mt + t) * 16 + half * 8;
            v8bf hv;
#pragma unroll
            for (int r = 0; r < 8; ++r) {
                float v = acc[r] + b1p[r];
                hv[r] = (__bf16)(v > 0.f ? v : 0.f);
            }
            *(v8bf*)(Hlds + m * HID + (mt + t) * 16 + half * 8) = hv;
        }
    }
    __syncthreads();

    // Stage 2: 16 M-tiles of 16 out channels; each wave does a pair sharing B.
    {
        int mt = 2 * widx;
        const __bf16* bcol2 = Hlds + m * HID + half * 8;
        const __bf16* arow0 = W2b + (size_t)(mt * 16 + m) * HID + half * 8;
        const __bf16* arow1 = arow0 + (size_t)16 * HID;
        v8f acc0 = {}, acc1 = {};
        for (int k = 0; k < HID; k += 32) {
            v16bf b  = load_fragB(bcol2, k);    // ds_load
            v16bf a0 = load_fragA(arow0, k);
            v16bf a1 = load_fragA(arow1, k);
            acc0 = wmma_bf16(a0, b, acc0);
            acc1 = wmma_bf16(a1, b, acc1);
        }
#pragma unroll
        for (int t = 0; t < 2; ++t) {
            v8f acc = t ? acc1 : acc0;
            float* op = pred + ((size_t)bb * HWP + p0 + m) * CD + (mt + t) * 16 + half * 8;
            const float* b2p = b2 + (mt + t) * 16 + half * 8;
#pragma unroll
            for (int r = 0; r < 8; ++r) op[r] = acc[r] + b2p[r];
        }
    }
}

// ---------------------------------------------------------------------------
// 7) Per-row cosine(pred, matched) over CD channels.
// ---------------------------------------------------------------------------
__global__ void k_rowcos(const float* __restrict__ pred, const float* __restrict__ matched,
                         float* __restrict__ rowcos) {
    int lane = threadIdx.x & 31;
    int row  = blockIdx.x * 8 + (threadIdx.x >> 5);
    int bb = row / HW, i = row % HW;
    const float* p = pred    + ((size_t)bb * HWP + i) * CD;
    const float* q = matched + ((size_t)bb * HW  + i) * CD;
    float dot = 0.f, np = 0.f, nq = 0.f;
    for (int c = lane; c < CD; c += 32) {
        float a = p[c], b = q[c];
        dot += a * b; np += a * a; nq += b * b;
    }
#pragma unroll
    for (int off = 16; off > 0; off >>= 1) {
        dot += __shfl_xor(dot, off, 32);
        np  += __shfl_xor(np,  off, 32);
        nq  += __shfl_xor(nq,  off, 32);
    }
    if (lane == 0)
        rowcos[row] = dot / fmaxf(sqrtf(np) * sqrtf(nq), 1e-8f);
}

// ---------------------------------------------------------------------------
// 8) Deterministic final reduction -> scalar loss.
// ---------------------------------------------------------------------------
__global__ void k_final(const float* __restrict__ rowcos, float* __restrict__ out) {
    __shared__ float s[256];
    float acc = 0.f;
    for (int i = threadIdx.x; i < BB * HW; i += 256) acc += rowcos[i];
    s[threadIdx.x] = acc;
    __syncthreads();
    for (int st = 128; st > 0; st >>= 1) {
        if (threadIdx.x < st) s[threadIdx.x] += s[threadIdx.x + st];
        __syncthreads();
    }
    if (threadIdx.x == 0) out[0] = -2.f * s[0] / (float)(BB * HW) + 2.f;
}

// ---------------------------------------------------------------------------
extern "C" void kernel_launch(void* const* d_in, const int* in_sizes, int n_in,
                              void* d_out, int out_size, void* d_ws, size_t ws_size,
                              hipStream_t stream) {
    const float* feat_on   = (const float*)d_in[0];
    const float* feat_targ = (const float*)d_in[1];
    const float* dense_on  = (const float*)d_in[2];
    const float* dense_tg  = (const float*)d_in[3];
    const float* W1        = (const float*)d_in[4];
    const float* b1        = (const float*)d_in[5];
    const float* W2        = (const float*)d_in[6];
    const float* b2        = (const float*)d_in[7];
    (void)in_sizes; (void)n_in; (void)out_size; (void)ws_size;

    char* w = (char*)d_ws;
    auto carve = [&](size_t bytes) -> void* {
        void* p = (void*)w;
        w += (bytes + 255) & ~(size_t)255;
        return p;
    };
    __bf16* A1      = (__bf16*)carve((size_t)BB * HWP * CF * 2);
    __bf16* A2      = (__bf16*)carve((size_t)BB * HWP * CF * 2);
    float*  inv1    = (float*) carve((size_t)BB * HW * 4);
    float*  inv2    = (float*) carve((size_t)BB * HW * 4);
    __bf16* W1b     = (__bf16*)carve((size_t)HID * CD * 2);
    __bf16* W2b     = (__bf16*)carve((size_t)CD * HID * 2);
    __bf16* xT      = (__bf16*)carve((size_t)BB * HWP * CD * 2);
    float*  d2T     = (float*) carve((size_t)BB * HW * CD * 4);
    float*  cosm    = (float*) carve((size_t)BB * HWP * HWP * 4);
    float*  matched = (float*) carve((size_t)BB * HW * CD * 4);
    float*  pred    = (float*) carve((size_t)BB * HWP * CD * 4);
    float*  rowcos  = (float*) carve((size_t)BB * HW * 4);

    k_cvt_w         <<<(HID * CD + 255) / 256, 256, 0, stream>>>(W1, W2, W1b, W2b);
    k_colnorm       <<<BB, 256, 0, stream>>>(feat_on, feat_targ, inv1, inv2);
    k_transpose_feat<<<dim3(NTI, BB), dim3(16, 16), 0, stream>>>(feat_on, feat_targ,
                                                                 inv1, inv2, A1, A2);
    k_prep_dense    <<<dim3(NTI, BB), dim3(16, 16), 0, stream>>>(dense_on, dense_tg, xT, d2T);
    k_gemm_cos      <<<(BB * NST * NST) / 8, 256, 0, stream>>>(A1, A2, cosm);
    k_argmax_gather <<<(BB * HW) / 8, 256, 0, stream>>>(cosm, d2T, matched);
    k_mlp           <<<dim3(NTI, BB), 256, 0, stream>>>(W1b, b1, W2b, b2, xT, pred);
    k_rowcos        <<<(BB * HW) / 8, 256, 0, stream>>>(pred, matched, rowcos);
    k_final         <<<1, 256, 0, stream>>>(rowcos, (float*)d_out);
}
